// EGNNResidueClassifier_61478161875532
// MI455X (gfx1250) — matrix operand order repaired and checked by reading, not compiled, not run on hardware
//
#include <hip/hip_runtime.h>

// ---------------------------------------------------------------------------
// EGNN residue classifier for MI455X (gfx1250), wave32, bf16 WMMA everywhere.
// ---------------------------------------------------------------------------

typedef __attribute__((ext_vector_type(16))) __bf16 v16bf;
typedef __attribute__((ext_vector_type(8)))  float  v8f;

#define NB    4      // batch
#define NNODE 256    // nodes per graph
#define HD    256    // hidden dim
#define NL    4      // layers

__device__ __forceinline__ float silu_f(float x){ return x / (1.0f + __expf(-x)); }
__device__ __forceinline__ float sigm_f(float x){ return 1.0f / (1.0f + __expf(-x)); }

__device__ __forceinline__ v8f zero8(){
  v8f z;
#pragma unroll
  for (int i = 0; i < 8; ++i) z[i] = 0.0f;
  return z;
}

union FragU { uint4 u[2]; v16bf v; };

// A fragment (16x32 bf16, MxK). Layout: lane<16 -> row=lane, K=colb+0..15
// (colb = kb*32); lane>=16 -> row=lane-16, colb += 16.
// Works for LDS or global base pointers; stride in elements (0 => broadcast row).
__device__ __forceinline__ v16bf load_a_frag(const __bf16* base, int stride, int lane, int kb){
  int row  = lane & 15;
  int colb = kb * 32 + ((lane & 16) ? 16 : 0);
  const __bf16* p = base + row * stride + colb;
  FragU r;
  r.u[0] = *(const uint4*)(p);
  r.u[1] = *(const uint4*)(p + 8);
  return r.v;
}

// B fragment (32x16 bf16, KxN) from pre-packed chunks: chunk = 1KB = 32 lanes x
// 16 contiguous bf16 (lane = column + K-half select), so 2 contiguous b128 loads.
__device__ __forceinline__ v16bf load_b_frag(const __bf16* pack, int chunk, int lane){
  const __bf16* p = pack + chunk * 512 + lane * 16;
  FragU r;
  r.u[0] = *(const uint4*)(p);
  r.u[1] = *(const uint4*)(p + 8);
  return r.v;
}

#define WMMA_BF16(A,B,C) \
  __builtin_amdgcn_wmma_f32_16x16x32_bf16(false, (A), false, (B), (short)0, (C), false, false)

// ---------------------------------------------------------------------------
// Weight packing: W [K x 256] f32 row-major -> bf16 B-fragment chunks.
// chunk c = (kb, nt): lane L covers col n = nt*16 + (L&15),
// K = kb*32 + (L<16 ? 0 : 16) + e, e in 0..15. k >= Krows padded with zero.
// ---------------------------------------------------------------------------
__global__ __launch_bounds__(256) void k_pack(const float* __restrict__ W, int Krows,
                                              int chunks, __bf16* __restrict__ dst){
  int g = blockIdx.x * 256 + threadIdx.x;
  if (g >= chunks * 32) return;
  int c    = g >> 5;
  int lane = g & 31;
  int kb = c >> 4, nt = c & 15;
  int n  = nt * 16 + (lane & 15);
  int k0 = kb * 32 + ((lane & 16) ? 16 : 0);
  __bf16* o = dst + c * 512 + lane * 16;
#pragma unroll
  for (int e = 0; e < 16; ++e){
    int k = k0 + e;
    float v = (k < Krows) ? W[k * 256 + n] : 0.0f;
    o[e] = (__bf16)v;
  }
}

// ---------------------------------------------------------------------------
// Embedding in: hid = h @ emb_in_w + b (K=64, trivial). Also copies x -> xcur.
// ---------------------------------------------------------------------------
__global__ __launch_bounds__(256) void k_embed(const float* __restrict__ h,
                                               const float* __restrict__ W,
                                               const float* __restrict__ bias,
                                               float* __restrict__ hid,
                                               __bf16* __restrict__ hidbf,
                                               const float* __restrict__ x,
                                               float* __restrict__ xcur){
  int row = blockIdx.x;        // 0..1023 (= B*N)
  int n   = threadIdx.x;       // 0..255
  float acc = bias[n];
#pragma unroll 8
  for (int k = 0; k < 64; ++k) acc += h[row * 64 + k] * W[k * 256 + n];
  hid  [row * 256 + n] = acc;
  hidbf[row * 256 + n] = (__bf16)acc;
  int xi = blockIdx.x * 256 + threadIdx.x;
  if (xi < NB * NNODE * 3) xcur[xi] = x[xi];
}

__global__ __launch_bounds__(256) void k_zero(float* __restrict__ agg,
                                              float* __restrict__ coordacc){
  int idx = blockIdx.x * 256 + threadIdx.x;   // grid covers exactly B*N*H
  agg[idx] = 0.0f;
  if (idx < NB * NNODE * 3) coordacc[idx] = 0.0f;
}

// ---------------------------------------------------------------------------
// Fused edge kernel. Block = 16-row i-tile x 8 j's (one j per wave).
// grid = B(4) * i-tiles(16) * j-chunks(32) = 2048 blocks, 256 threads.
// ---------------------------------------------------------------------------
#define SSTR 264   // stage/tile row stride in bf16 elements (bank-conflict pad)

__global__ __launch_bounds__(256) void k_edge(
    const __bf16* __restrict__ hidbf, const float* __restrict__ xcur,
    float* __restrict__ aggacc, float* __restrict__ coordacc,
    const __bf16* __restrict__ pW1, const __bf16* __restrict__ pW2,
    const __bf16* __restrict__ pW3,
    const float* __restrict__ w1tail,   // edge_w1 rows 512 (radial) & 513 (ones)
    const float* __restrict__ eb1, const float* __restrict__ eb2,
    const float* __restrict__ attw, const float* __restrict__ attb,
    const float* __restrict__ cb1, const float* __restrict__ cw2)
{
  extern __shared__ char smraw[];
  __bf16* sHi    = (__bf16*)smraw;          // 16 x SSTR
  __bf16* sHj    = sHi + 16 * SSTR;         //  8 x SSTR
  __bf16* sStage = sHj + 8 * SSTR;          //  8 waves x 16 x SSTR
  float*  sAgg   = (float*)(smraw + 80256); // 16 x 256
  float*  sCoord = sAgg + 16 * 256;         // 48
  float*  sXi    = sCoord + 48;             // 48
  float*  sXj    = sXi + 48;                // 32 (24 used)
  float*  sGate  = sXj + 32;                // 8 x 16

  int tid = threadIdx.x, lane = tid & 31, wv = tid >> 5;
  int bid = blockIdx.x;
  int jc = bid & 31, it = (bid >> 5) & 15, b = bid >> 9;
  int i0 = it * 16, j0 = jc * 8;
  int jIdx = j0 + wv;

  // ---- cooperative tile loads + zero accumulators ----
  for (int t = tid; t < 512; t += 256){                 // hi: 16 rows x 256 bf16
    int r = t >> 5; int n = (t & 31) * 8;
    *(uint4*)(sHi + r * SSTR + n) = *(const uint4*)(hidbf + ((b * NNODE + i0 + r) * HD) + n);
  }
  {
    int t = tid;
    if (t < 256){                                       // hj: 8 rows x 256 bf16
      int r = t >> 5; int n = (t & 31) * 8;
      *(uint4*)(sHj + r * SSTR + n) = *(const uint4*)(hidbf + ((b * NNODE + j0 + r) * HD) + n);
    }
  }
  if (tid < 48) sXi[tid] = xcur[(b * NNODE + i0) * 3 + tid];
  if (tid >= 64 && tid < 88) sXj[tid - 64] = xcur[(b * NNODE + j0) * 3 + (tid - 64)];
  for (int t = tid; t < 16 * 256; t += 256) sAgg[t] = 0.0f;
  if (tid < 48) sCoord[tid] = 0.0f;
  __syncthreads();

  __bf16* stg = sStage + wv * (16 * SSTR);
  const __bf16* hjRow = sHj + wv * SSTR;
  int h8 = (lane & 16) ? 8 : 0;
  int nlane = lane & 15;
  float xj0 = sXj[wv * 3 + 0], xj1 = sXj[wv * 3 + 1], xj2 = sXj[wv * 3 + 2];

  float radf[8];
#pragma unroll
  for (int v = 0; v < 8; ++v){
    int r = v + h8;
    float d0 = sXi[r * 3 + 0] - xj0, d1 = sXi[r * 3 + 1] - xj1, d2 = sXi[r * 3 + 2] - xj2;
    radf[v] = d0 * d0 + d1 * d1 + d2 * d2;
  }

  // ================= GEMM1: feat[16x514] @ W1 -> m1 [16x256] ================
  v8f C[16];
#pragma unroll
  for (int nt = 0; nt < 16; ++nt) C[nt] = zero8();
  for (int kb = 0; kb < 16; ++kb){
    __builtin_prefetch((const char*)pW1 + (kb + 1) * 16384, 0, 1);
    v16bf A = (kb < 8) ? load_a_frag(sHi, SSTR, lane, kb)
                       : load_a_frag(hjRow, 0, lane, kb - 8);
#pragma unroll
    for (int nt = 0; nt < 16; ++nt){
      v16bf Bf = load_b_frag(pW1, kb * 16 + nt, lane);
      C[nt] = WMMA_BF16(A, Bf, C[nt]);
    }
  }
  // epilogue: + radial*W1[512] + W1[513] + b1, SiLU, stage as bf16
#pragma unroll
  for (int nt = 0; nt < 16; ++nt){
    int n = nt * 16 + nlane;
    float w512 = w1tail[n];
    float w513 = w1tail[256 + n];
    float bb   = eb1[n];
#pragma unroll
    for (int v = 0; v < 8; ++v){
      float t = C[nt][v] + radf[v] * w512 + w513 + bb;
      stg[(v + h8) * SSTR + n] = (__bf16)silu_f(t);
    }
  }

  // ================= GEMM2: m1 @ W2 -> m [16x256] + attention ===============
#pragma unroll
  for (int nt = 0; nt < 16; ++nt) C[nt] = zero8();
  for (int kb = 0; kb < 8; ++kb){
    __builtin_prefetch((const char*)pW2 + (kb + 1) * 16384, 0, 1);
    v16bf A = load_a_frag(stg, SSTR, lane, kb);
#pragma unroll
    for (int nt = 0; nt < 16; ++nt){
      v16bf Bf = load_b_frag(pW2, kb * 16 + nt, lane);
      C[nt] = WMMA_BF16(A, Bf, C[nt]);
    }
  }
  float attp[8];
#pragma unroll
  for (int v = 0; v < 8; ++v) attp[v] = 0.0f;
#pragma unroll
  for (int nt = 0; nt < 16; ++nt){
    int n = nt * 16 + nlane;
    float aw = attw[n];
    float bb = eb2[n];
#pragma unroll
    for (int v = 0; v < 8; ++v){
      float m = silu_f(C[nt][v] + bb);
      attp[v] += m * aw;
      stg[(v + h8) * SSTR + n] = (__bf16)m;   // overwrite stage with m (pre-gate)
    }
  }
  // gate = sigmoid(m @ att_w + att_b); zero the diagonal edge (i == j)
#pragma unroll
  for (int v = 0; v < 8; ++v){
    attp[v] += __shfl_xor(attp[v], 1);
    attp[v] += __shfl_xor(attp[v], 2);
    attp[v] += __shfl_xor(attp[v], 4);
    attp[v] += __shfl_xor(attp[v], 8);
  }
  float ab = attb[0];
  if (nlane == 0){
#pragma unroll
    for (int v = 0; v < 8; ++v){
      int r = v + h8;
      float g = sigm_f(attp[v] + ab);
      if (i0 + r == jIdx) g = 0.0f;
      sGate[wv * 16 + r] = g;
    }
  }
  __builtin_amdgcn_wave_barrier();

  // scale stage by gate and accumulate agg into block-shared LDS
  {
    float g[16];
#pragma unroll
    for (int r = 0; r < 16; ++r) g[r] = sGate[wv * 16 + r];
    for (int idx = lane; idx < 4096; idx += 32){
      int r = idx >> 8, n = idx & 255;
      float mv = (float)stg[r * SSTR + n] * g[r];
      stg[r * SSTR + n] = (__bf16)mv;
      atomicAdd(&sAgg[r * 256 + n], mv);
    }
  }

  // ============ GEMM3: m_gated @ coord_w1, SiLU, dot coord_w2 ===============
  float cwp[8];
#pragma unroll
  for (int v = 0; v < 8; ++v) cwp[v] = 0.0f;
  for (int nt = 0; nt < 16; ++nt){
    v8f C3 = zero8();
    for (int kb = 0; kb < 8; ++kb){
      v16bf A  = load_a_frag(stg, SSTR, lane, kb);
      v16bf Bf = load_b_frag(pW3, kb * 16 + nt, lane);
      C3 = WMMA_BF16(A, Bf, C3);
    }
    int n = nt * 16 + nlane;
    float b1v = cb1[n];
    float w2v = cw2[n];
#pragma unroll
    for (int v = 0; v < 8; ++v) cwp[v] += silu_f(C3[v] + b1v) * w2v;
  }
#pragma unroll
  for (int v = 0; v < 8; ++v){
    cwp[v] += __shfl_xor(cwp[v], 1);
    cwp[v] += __shfl_xor(cwp[v], 2);
    cwp[v] += __shfl_xor(cwp[v], 4);
    cwp[v] += __shfl_xor(cwp[v], 8);
  }
  if (nlane == 0){
#pragma unroll
    for (int v = 0; v < 8; ++v){
      int r = v + h8;
      float cw = cwp[v];
      atomicAdd(&sCoord[r * 3 + 0], (sXi[r * 3 + 0] - xj0) * cw);
      atomicAdd(&sCoord[r * 3 + 1], (sXi[r * 3 + 1] - xj1) * cw);
      atomicAdd(&sCoord[r * 3 + 2], (sXi[r * 3 + 2] - xj2) * cw);
    }
  }
  __syncthreads();

  // ---- flush block partials to global ----
  for (int t = tid; t < 4096; t += 256){
    int r = t >> 8, n = t & 255;
    atomicAdd(aggacc + (b * NNODE + i0 + r) * HD + n, sAgg[t]);
  }
  if (tid < 48) atomicAdd(coordacc + (b * NNODE + i0) * 3 + tid, sCoord[tid]);
}

// ---------------------------------------------------------------------------
// x += coordacc / (N-1); agg f32 -> bf16
// ---------------------------------------------------------------------------
__global__ __launch_bounds__(256) void k_prep(const float* __restrict__ agg,
                                              __bf16* __restrict__ aggbf,
                                              float* __restrict__ xcur,
                                              const float* __restrict__ coordacc){
  int idx = blockIdx.x * 256 + threadIdx.x;
  aggbf[idx] = (__bf16)agg[idx];
  if (idx < NB * NNODE * 3) xcur[idx] += coordacc[idx] * (1.0f / 255.0f);
}

// ---------------------------------------------------------------------------
// Node MLP with residual: hid += silu([hid,agg]@W1+b1)@W2 + b2   (WMMA)
// grid = 8 blocks x 8 waves; wave = one 16-row M-tile of the 1024 rows.
// ---------------------------------------------------------------------------
__global__ __launch_bounds__(256) void k_node(float* __restrict__ hid,
                                              __bf16* __restrict__ hidbf,
                                              const __bf16* __restrict__ aggbf,
                                              const __bf16* __restrict__ pNW1,
                                              const __bf16* __restrict__ pNW2,
                                              const float* __restrict__ nb1,
                                              const float* __restrict__ nb2){
  extern __shared__ char smraw[];
  __bf16* sStage = (__bf16*)smraw;
  int tid = threadIdx.x, lane = tid & 31, wv = tid >> 5;
  int mt = blockIdx.x * 8 + wv;
  int r0 = mt * 16;
  __bf16* stg = sStage + wv * (16 * SSTR);
  int h8 = (lane & 16) ? 8 : 0, nlane = lane & 15;

  v8f C[16];
#pragma unroll
  for (int nt = 0; nt < 16; ++nt) C[nt] = zero8();
  for (int kb = 0; kb < 16; ++kb){
    __builtin_prefetch((const char*)pNW1 + (kb + 1) * 16384, 0, 1);
    v16bf A = (kb < 8) ? load_a_frag(hidbf + r0 * HD, HD, lane, kb)
                       : load_a_frag(aggbf + r0 * HD, HD, lane, kb - 8);
#pragma unroll
    for (int nt = 0; nt < 16; ++nt){
      v16bf Bf = load_b_frag(pNW1, kb * 16 + nt, lane);
      C[nt] = WMMA_BF16(A, Bf, C[nt]);
    }
  }
#pragma unroll
  for (int nt = 0; nt < 16; ++nt){
    int n = nt * 16 + nlane;
    float bb = nb1[n];
#pragma unroll
    for (int v = 0; v < 8; ++v)
      stg[(v + h8) * SSTR + n] = (__bf16)silu_f(C[nt][v] + bb);
  }

#pragma unroll
  for (int nt = 0; nt < 16; ++nt) C[nt] = zero8();
  for (int kb = 0; kb < 8; ++kb){
    v16bf A = load_a_frag(stg, SSTR, lane, kb);
#pragma unroll
    for (int nt = 0; nt < 16; ++nt){
      v16bf Bf = load_b_frag(pNW2, kb * 16 + nt, lane);
      C[nt] = WMMA_BF16(A, Bf, C[nt]);
    }
  }
#pragma unroll
  for (int nt = 0; nt < 16; ++nt){
    int n = nt * 16 + nlane;
    float bb = nb2[n];
#pragma unroll
    for (int v = 0; v < 8; ++v){
      int row = r0 + v + h8;
      float o = hid[row * HD + n] + C[nt][v] + bb;
      hid  [row * HD + n] = o;
      hidbf[row * HD + n] = (__bf16)o;
    }
  }
}

// ---------------------------------------------------------------------------
// Head: mean-pool commutes with linear emb_out -> tiny sequential GEMMs.
// ---------------------------------------------------------------------------
__global__ __launch_bounds__(256) void k_head(const float* __restrict__ hid,
    const float* __restrict__ eow, const float* __restrict__ eob,
    const float* __restrict__ hw1, const float* __restrict__ hb1,
    const float* __restrict__ hw2, const float* __restrict__ hb2,
    float* __restrict__ out){
  __shared__ float sp[4 * 256];
  __shared__ float se[4 * 256];
  int tid = threadIdx.x;
  for (int t = tid; t < 1024; t += 256){
    int b = t >> 8, n = t & 255;
    float s = 0.0f;
    for (int i = 0; i < 256; ++i) s += hid[(b * 256 + i) * 256 + n];
    sp[t] = s * (1.0f / 256.0f);
  }
  __syncthreads();
  for (int t = tid; t < 1024; t += 256){
    int b = t >> 8, n = t & 255;
    float a = eob[n];
    for (int k = 0; k < 256; ++k) a += sp[b * 256 + k] * eow[k * 256 + n];
    se[t] = a;
  }
  __syncthreads();
  for (int t = tid; t < 1024; t += 256){
    int b = t >> 8, n = t & 255;
    float a = hb1[n];
    for (int k = 0; k < 256; ++k) a += se[b * 256 + k] * hw1[k * 256 + n];
    sp[t] = fmaxf(a, 0.0f);
  }
  __syncthreads();
  if (tid < 80){
    int b = tid / 20, o = tid % 20;
    float a = hb2[o];
    for (int k = 0; k < 256; ++k) a += sp[b * 256 + k] * hw2[k * 20 + o];
    out[tid] = a;
  }
}

// ---------------------------------------------------------------------------
// Host launcher
// ---------------------------------------------------------------------------
#define EDGE_LDS 97664
#define NODE_LDS 67584

extern "C" void kernel_launch(void* const* d_in, const int* in_sizes, int n_in,
                              void* d_out, int out_size, void* d_ws, size_t ws_size,
                              hipStream_t stream){
  const float* h        = (const float*)d_in[0];
  const float* x        = (const float*)d_in[1];
  const float* emb_in_w = (const float*)d_in[2];
  const float* emb_in_b = (const float*)d_in[3];
  const float* edge_w1  = (const float*)d_in[4];
  const float* edge_b1  = (const float*)d_in[5];
  const float* edge_w2  = (const float*)d_in[6];
  const float* edge_b2  = (const float*)d_in[7];
  const float* att_w    = (const float*)d_in[8];
  const float* att_b    = (const float*)d_in[9];
  const float* coord_w1 = (const float*)d_in[10];
  const float* coord_b1 = (const float*)d_in[11];
  const float* coord_w2 = (const float*)d_in[12];
  const float* node_w1  = (const float*)d_in[13];
  const float* node_b1  = (const float*)d_in[14];
  const float* node_w2  = (const float*)d_in[15];
  const float* node_b2  = (const float*)d_in[16];
  const float* emb_out_w= (const float*)d_in[17];
  const float* emb_out_b= (const float*)d_in[18];
  const float* head_w1  = (const float*)d_in[19];
  const float* head_b1  = (const float*)d_in[20];
  const float* head_w2  = (const float*)d_in[21];
  const float* head_b2  = (const float*)d_in[22];

  char* ws = (char*)d_ws;
  float*  hid      = (float*) (ws + 0);
  __bf16* hidbf    = (__bf16*)(ws + 1048576);
  float*  agg      = (float*) (ws + 1572864);
  __bf16* aggbf    = (__bf16*)(ws + 2621440);
  float*  coordacc = (float*) (ws + 3145728);
  float*  xcur     = (float*) (ws + 3158016);
  __bf16* pack     = (__bf16*)(ws + 3170304);

  const size_t LSTRIDE = 458752;   // bf16 elems of packed weights per layer
  // per-layer packed sub-offsets (bf16 elems)
  const size_t OW1 = 0, OW2 = 131072, OW3 = 196608, ON1 = 262144, ON2 = 393216;

  // pack all GEMM weights into WMMA B-fragment layout (every call; deterministic)
  for (int l = 0; l < NL; ++l){
    __bf16* pb = pack + l * LSTRIDE;
    k_pack<<<32, 256, 0, stream>>>(edge_w1 + (size_t)l * 514 * 256, 512, 256, pb + OW1);
    k_pack<<<16, 256, 0, stream>>>(edge_w2 + (size_t)l * 65536,     256, 128, pb + OW2);
    k_pack<<<16, 256, 0, stream>>>(coord_w1 + (size_t)l * 65536,    256, 128, pb + OW3);
    k_pack<<<32, 256, 0, stream>>>(node_w1 + (size_t)l * 131072,    512, 256, pb + ON1);
    k_pack<<<16, 256, 0, stream>>>(node_w2 + (size_t)l * 65536,     256, 128, pb + ON2);
  }

  k_embed<<<1024, 256, 0, stream>>>(h, emb_in_w, emb_in_b, hid, hidbf, x, xcur);

  for (int l = 0; l < NL; ++l){
    __bf16* pb = pack + l * LSTRIDE;
    k_zero<<<1024, 256, 0, stream>>>(agg, coordacc);
    k_edge<<<2048, 256, EDGE_LDS, stream>>>(
        hidbf, xcur, agg, coordacc,
        pb + OW1, pb + OW2, pb + OW3,
        edge_w1 + (size_t)l * 514 * 256 + 512 * 256,   // rows 512/513
        edge_b1 + l * 256, edge_b2 + l * 256,
        att_w + l * 256, att_b + l,
        coord_b1 + l * 256, coord_w2 + l * 256);
    k_prep<<<1024, 256, 0, stream>>>(agg, aggbf, xcur, coordacc);
    k_node<<<8, 256, NODE_LDS, stream>>>(hid, hidbf, aggbf,
        pb + ON1, pb + ON2, node_b1 + l * 256, node_b2 + l * 256);
  }

  k_head<<<1, 256, 0, stream>>>(hid, emb_out_w, emb_out_b,
                                head_w1, head_b1, head_w2, head_b2,
                                (float*)d_out);
}
